// ResidualGATLayer_41386304864862
// MI455X (gfx1250) — compile-verified, matrix operand order checked by reading, not run on hardware
//
#include <hip/hip_runtime.h>
#include <hip/hip_bf16.h>
#include <stdint.h>

// Problem constants (from reference setup_inputs)
#define HID   256
#define NHEAD 4
#define CPH   64          // HID / NHEAD
#define NEG_SLOPE 0.2f
#define EPS_SM    1e-16f
#define LN_EPS    1e-5f

typedef __attribute__((ext_vector_type(16))) __bf16          v16bf;
typedef __attribute__((ext_vector_type(16))) unsigned short  v16u;
typedef __attribute__((ext_vector_type(8)))  float           v8f;

// ---------- helpers ----------
__device__ __forceinline__ unsigned short f2bf(float f) {
  unsigned u = __float_as_uint(f);
  u += 0x7FFFu + ((u >> 16) & 1u);      // round-to-nearest-even truncation
  return (unsigned short)(u >> 16);
}
// order-preserving float<->uint encoding for atomicMax on floats
__device__ __forceinline__ unsigned encf(float f) {
  unsigned u = __float_as_uint(f);
  return (u & 0x80000000u) ? ~u : (u | 0x80000000u);
}
__device__ __forceinline__ float decf(unsigned e) {
  unsigned v = (e & 0x80000000u) ? (e & 0x7FFFFFFFu) : ~e;
  return __uint_as_float(v);
}

// ---------- 1) degree + incoming edge_attr sum (for self-loop fill 'mean') ----------
__global__ void degree_kernel(const int* __restrict__ dst,
                              const float* __restrict__ eattr,
                              float* __restrict__ deg,
                              float* __restrict__ loop_sum, int E) {
  int e = blockIdx.x * blockDim.x + threadIdx.x;
  if (e >= E) return;
  int d = dst[e];
  atomicAdd(&deg[d], 1.0f);
  atomicAdd(&loop_sum[d], eattr[e]);
}

// ---------- 2) we_dot[h] = sum_c We[h*64+c] * att_edge[h,c] ----------
__global__ void wedot_kernel(const float* __restrict__ We,
                             const float* __restrict__ att_edge,
                             float* __restrict__ wedot) {
  int h = threadIdx.x;
  if (h < NHEAD) {
    float s = 0.f;
    for (int c = 0; c < CPH; ++c) s += We[h * CPH + c] * att_edge[h * CPH + c];
    wedot[h] = s;
  }
}

// ---------- 3) convert x -> bf16 rowmajor (b128 in, b64 out), W -> bf16 transposed ----------
__global__ void conv_x_kernel(const float* __restrict__ x,
                              unsigned short* __restrict__ xbf, int total) {
  int i = (blockIdx.x * blockDim.x + threadIdx.x) * 4;
  if (i >= total) return;
  const float4 v = *(const float4*)(x + i);
  ushort4 o;
  o.x = f2bf(v.x); o.y = f2bf(v.y); o.z = f2bf(v.z); o.w = f2bf(v.w);
  *(ushort4*)(xbf + i) = o;
}
__global__ void conv_wt_kernel(const float* __restrict__ W,
                               unsigned short* __restrict__ wtbf) {
  int i = blockIdx.x * blockDim.x + threadIdx.x;
  if (i < HID * HID) {
    int k = i / HID, j = i % HID;
    wtbf[j * HID + k] = f2bf(W[k * HID + j]);   // Wt[j,k] = W[k,j]
  }
}

// ---------- 4) xp = x @ W via v_wmma_f32_16x16x32_bf16 ----------
// One wave computes a 16(M) x 64(N) strip: 4 accumulators, A fragment loaded
// once per K-step and reused across 4 WMMAs (4x less A-fragment L2 traffic).
// Fragment layouts per CDNA5 ISA 7.12.2:
//   A 16x32 bf16: lanes 0-15 row=lane, K in {0..7,16..23}; lanes 16-31 K in {8..15,24..31}
//   B 32x16 bf16: mirrored with N as the lane index (B loaded from W^T rows)
//   C/D 16x16 f32: vgpr r -> (m=r, n=lane) lanes 0-15; (m=r+8, n=lane-16) lanes 16-31
__global__ void gemm_wmma_kernel(const unsigned short* __restrict__ xbf,   // [N,256]
                                 const unsigned short* __restrict__ wtbf,  // [256,256] = W^T
                                 float* __restrict__ xp, int N) {
  const int lane = threadIdx.x & 31;
  const int wave = threadIdx.x >> 5;
  const int tile = blockIdx.x * 8 + wave;       // strip index
  const int stripsN = HID / 64;                 // 4 strips of 64 columns
  const int tm  = tile / stripsN;
  const int tn4 = tile % stripsN;
  if (tm * 16 >= N) return;
  const int half = lane >> 4;
  const int l15  = lane & 15;
  const int row  = tm * 16 + l15;
  const int arow = row < N ? row : N - 1;       // clamp only matters if N%16 != 0
  const int kofs = half ? 8 : 0;
  const int colb = tn4 * 64 + l15;              // base column; B tiles at +0,+16,+32,+48

  v8f acc[4] = {};
  for (int k0 = 0; k0 < HID; k0 += 32) {
    v16u au;
    const unsigned short* ap = xbf + arow * HID + k0 + kofs;
#pragma unroll
    for (int i = 0; i < 8; ++i) { au[i] = ap[i]; au[i + 8] = ap[i + 16]; }
    v16bf a = __builtin_bit_cast(v16bf, au);
#pragma unroll
    for (int j = 0; j < 4; ++j) {
      v16u bu;
      const unsigned short* bp = wtbf + (colb + j * 16) * HID + k0 + kofs;
#pragma unroll
      for (int i = 0; i < 8; ++i) { bu[i] = bp[i]; bu[i + 8] = bp[i + 16]; }
      v16bf b = __builtin_bit_cast(v16bf, bu);
      acc[j] = __builtin_amdgcn_wmma_f32_16x16x32_bf16(false, a, false, b,
                                                       (short)0, acc[j], false, false);
    }
  }
  const int mbase = tm * 16 + (half ? 8 : 0);
#pragma unroll
  for (int j = 0; j < 4; ++j) {
#pragma unroll
    for (int r = 0; r < 8; ++r) {
      int m = mbase + r;
      if (m < N) xp[m * HID + colb + j * 16] = acc[j][r];
    }
  }
}

// ---------- 5) a_s / a_d per (node, head) ----------
__global__ void attn_kernel(const float* __restrict__ xp,
                            const float* __restrict__ att_src,
                            const float* __restrict__ att_dst,
                            float* __restrict__ a_s, float* __restrict__ a_d, int N) {
  int i = blockIdx.x * blockDim.x + threadIdx.x;
  if (i >= N * NHEAD) return;
  int n = i / NHEAD, h = i % NHEAD;
  const float* v = xp + n * HID + h * CPH;
  float ss = 0.f, sd = 0.f;
  for (int c = 0; c < CPH; ++c) {
    float xv = v[c];
    ss += xv * att_src[h * CPH + c];
    sd += xv * att_dst[h * CPH + c];
  }
  a_s[i] = ss;
  a_d[i] = sd;
}

// ---------- 6a) edge pass: logits + segment max ----------
__global__ void edge_max_kernel(const int* __restrict__ src, const int* __restrict__ dst,
                                const float* __restrict__ eattr,
                                const float* __restrict__ deg,
                                const float* __restrict__ loop_sum,
                                const float* __restrict__ a_s, const float* __restrict__ a_d,
                                const float* __restrict__ wedot,
                                float* __restrict__ logits, unsigned* __restrict__ maxenc,
                                int E, int T) {
  int e = blockIdx.x * blockDim.x + threadIdx.x;
  if (e >= T) return;
  int s, d; float at;
  if (e < E) { s = src[e]; d = dst[e]; at = eattr[e]; }
  else       { s = d = e - E; at = loop_sum[s] / fmaxf(deg[s], 1.0f); }
#pragma unroll
  for (int h = 0; h < NHEAD; ++h) {
    float l = a_s[s * NHEAD + h] + a_d[d * NHEAD + h] + at * wedot[h];
    l = (l > 0.f) ? l : NEG_SLOPE * l;          // leaky_relu
    logits[e * NHEAD + h] = l;
    atomicMax(&maxenc[d * NHEAD + h], encf(l));
  }
}

// ---------- 6b) edge pass: exp + segment sum (logits buffer becomes p) ----------
__global__ void edge_sum_kernel(const int* __restrict__ dst,
                                const unsigned* __restrict__ maxenc,
                                float* __restrict__ logits,   // in logits / out p
                                float* __restrict__ expsum, int E, int T) {
  int e = blockIdx.x * blockDim.x + threadIdx.x;
  if (e >= T) return;
  int d = (e < E) ? dst[e] : (e - E);
#pragma unroll
  for (int h = 0; h < NHEAD; ++h) {
    float m = decf(maxenc[d * NHEAD + h]);
    float p = __expf(logits[e * NHEAD + h] - m);
    logits[e * NHEAD + h] = p;
    atomicAdd(&expsum[d * NHEAD + h], p);
  }
}

// ---------- 6c) edge pass: alpha-weighted gather-scatter ----------
// 4 edges per 256-thread block; each lane handles 4 consecutive features of one
// edge (b128 gather of xp[src]); head index is constant across the 4 features.
__global__ void edge_scatter_kernel(const int* __restrict__ src, const int* __restrict__ dst,
                                    const float* __restrict__ p,
                                    const float* __restrict__ expsum,
                                    const float* __restrict__ xp,
                                    float* __restrict__ agg, int E, int T) {
  int e = blockIdx.x * 4 + (threadIdx.x >> 6);
  if (e >= T) return;
  int t = (threadIdx.x & 63) * 4;               // feature base (4-aligned, head-uniform)
  int h = t >> 6;
  int s, d;
  if (e < E) { s = src[e]; d = dst[e]; }
  else       { s = d = e - E; }
  float alpha = p[e * NHEAD + h] / (expsum[d * NHEAD + h] + EPS_SM);
  const float4 xv = *(const float4*)(xp + s * HID + t);
  float* ag = agg + d * HID + t;
  atomicAdd(ag + 0, xv.x * alpha);
  atomicAdd(ag + 1, xv.y * alpha);
  atomicAdd(ag + 2, xv.z * alpha);
  atomicAdd(ag + 3, xv.w * alpha);
}

// ---------- 7) bias + LayerNorm + residual + ReLU ----------
__global__ void finalize_kernel(const float* __restrict__ agg, const float* __restrict__ x,
                                const float* __restrict__ bias,
                                const float* __restrict__ gamma, const float* __restrict__ beta,
                                float* __restrict__ out, int N) {
  __shared__ float red[HID];
  int n = blockIdx.x, t = threadIdx.x;
  float v = agg[n * HID + t] + bias[t];
  red[t] = v; __syncthreads();
  for (int s = HID / 2; s > 0; s >>= 1) {
    if (t < s) red[t] += red[t + s];
    __syncthreads();
  }
  float mu = red[0] * (1.0f / HID);
  __syncthreads();
  float dv = v - mu;
  red[t] = dv * dv; __syncthreads();
  for (int s = HID / 2; s > 0; s >>= 1) {
    if (t < s) red[t] += red[t + s];
    __syncthreads();
  }
  float var = red[0] * (1.0f / HID);
  float hn = dv * rsqrtf(var + LN_EPS) * gamma[t] + beta[t];
  float r = hn + x[n * HID + t];
  out[n * HID + t] = r > 0.f ? r : 0.f;
}

// ---------------- host launcher ----------------
extern "C" void kernel_launch(void* const* d_in, const int* in_sizes, int n_in,
                              void* d_out, int out_size, void* d_ws, size_t ws_size,
                              hipStream_t stream) {
  const float* x        = (const float*)d_in[0];
  const int*   eidx     = (const int*)  d_in[1];
  const float* eattr    = (const float*)d_in[2];
  const float* W        = (const float*)d_in[3];
  const float* att_src  = (const float*)d_in[4];
  const float* att_dst  = (const float*)d_in[5];
  const float* We       = (const float*)d_in[6];
  const float* att_edge = (const float*)d_in[7];
  const float* bias     = (const float*)d_in[8];
  const float* gamma    = (const float*)d_in[9];
  const float* beta     = (const float*)d_in[10];

  const int N = in_sizes[0] / HID;
  const int E = in_sizes[1] / 2;
  const int T = E + N;                       // edges incl. self loops
  const int* src = eidx;
  const int* dst = eidx + E;

  // --- carve workspace (256B aligned) ---
  char* base = (char*)d_ws;
  size_t off = 0;
  auto carve = [&](size_t bytes) -> char* {
    char* p = base + off;
    off = (off + bytes + 255) & ~(size_t)255;
    return p;
  };
  float*          deg      = (float*)carve((size_t)N * 4);
  float*          loop_sum = (float*)carve((size_t)N * 4);
  unsigned short* xbf      = (unsigned short*)carve((size_t)N * HID * 2);
  unsigned short* wtbf     = (unsigned short*)carve((size_t)HID * HID * 2);
  float*          xp       = (float*)carve((size_t)N * HID * 4);
  float*          a_s      = (float*)carve((size_t)N * NHEAD * 4);
  float*          a_d      = (float*)carve((size_t)N * NHEAD * 4);
  float*          wedot    = (float*)carve(NHEAD * 4);
  unsigned*       maxenc   = (unsigned*)carve((size_t)N * NHEAD * 4);
  float*          expsum   = (float*)carve((size_t)N * NHEAD * 4);
  float*          pbuf     = (float*)carve((size_t)T * NHEAD * 4);
  float*          agg      = (float*)carve((size_t)N * HID * 4);

  // --- zero accumulators (graph-capture-safe async memsets) ---
  hipMemsetAsync(deg,      0, (size_t)N * 4, stream);
  hipMemsetAsync(loop_sum, 0, (size_t)N * 4, stream);
  hipMemsetAsync(maxenc,   0, (size_t)N * NHEAD * 4, stream);  // 0 == encoded minimum
  hipMemsetAsync(expsum,   0, (size_t)N * NHEAD * 4, stream);
  hipMemsetAsync(agg,      0, (size_t)N * HID * 4, stream);

  // 1) degree / loop attr
  degree_kernel<<<(E + 255) / 256, 256, 0, stream>>>(dst, eattr, deg, loop_sum, E);

  // 2) we_dot
  wedot_kernel<<<1, 32, 0, stream>>>(We, att_edge, wedot);

  // 3) precision conversion for WMMA
  conv_x_kernel <<<(((size_t)N * HID / 4) + 255) / 256, 256, 0, stream>>>(x, xbf, N * HID);
  conv_wt_kernel<<<(HID * HID + 255) / 256, 256, 0, stream>>>(W, wtbf);

  // 4) xp = x @ W  (bf16 WMMA, f32 accumulate, 16x64 strip per wave)
  {
    int tilesM = (N + 15) / 16;
    int strips = tilesM * (HID / 64);
    gemm_wmma_kernel<<<(strips + 7) / 8, 256, 0, stream>>>(xbf, wtbf, xp, N);
  }

  // 5) a_s, a_d
  attn_kernel<<<(N * NHEAD + 255) / 256, 256, 0, stream>>>(xp, att_src, att_dst, a_s, a_d, N);

  // 6) edge softmax passes
  edge_max_kernel<<<(T + 255) / 256, 256, 0, stream>>>(src, dst, eattr, deg, loop_sum,
                                                       a_s, a_d, wedot, pbuf, maxenc, E, T);
  edge_sum_kernel<<<(T + 255) / 256, 256, 0, stream>>>(dst, maxenc, pbuf, expsum, E, T);
  edge_scatter_kernel<<<(T + 3) / 4, 256, 0, stream>>>(src, dst, pbuf, expsum, xp, agg, E, T);

  // 7) LayerNorm + residual + ReLU
  finalize_kernel<<<N, 256, 0, stream>>>(agg, x, bias, gamma, beta, (float*)d_out, N);
}
